// TemporalInitialPoseModel_6760278524533
// MI455X (gfx1250) — matrix operand order repaired and checked by reading, not compiled, not run on hardware
//
#include <hip/hip_runtime.h>
#include <math.h>

// Problem constants (from setup_inputs)
#define BATCH 64
#define NPTS  12288
#define NKP   9        // 8 keypoint offsets + 1 center offset
#define KCAND 10
#define BIGF  1000000.0f
#define CHUNK 256
#define NCHUNK (NPTS / CHUNK)   // 48, exact

typedef __attribute__((ext_vector_type(16))) _Float16 v16h;
typedef __attribute__((ext_vector_type(8)))  float    v8f;

#if __has_builtin(__builtin_amdgcn_global_load_async_to_lds_b32) && \
    __has_builtin(__builtin_amdgcn_global_load_async_to_lds_b64) && \
    __has_builtin(__builtin_amdgcn_s_wait_asynccnt)
#define USE_ASYNC_LDS 1

// Builtin signatures (from hipcc diagnostics):
//   b32: (addrspace(1) int*,            addrspace(3) int*,            imm offset, imm cpol)
//   b64: (addrspace(1) int vec2*,       addrspace(3) int vec2*,       imm offset, imm cpol)
typedef int v2i_t __attribute__((__vector_size__(8)));
typedef __attribute__((address_space(1))) int   gint_t;
typedef __attribute__((address_space(1))) v2i_t gv2i_t;
typedef __attribute__((address_space(3))) int   lint_t;
typedef __attribute__((address_space(3))) v2i_t lv2i_t;

// Direct pointer casts across address spaces are ill-formed; go through ints.
// Global generic pointers are identity-mapped to addrspace(1).
__device__ __forceinline__ gint_t* as_global_i32(const void* p) {
    return (gint_t*)(unsigned long long)p;
}
__device__ __forceinline__ gv2i_t* as_global_v2i(const void* p) {
    return (gv2i_t*)(unsigned long long)p;
}
// Generic LDS pointer low 32 bits == LDS offset.
__device__ __forceinline__ lint_t* as_lds_i32(void* p) {
    return (lint_t*)(unsigned int)(unsigned long long)p;
}
__device__ __forceinline__ lv2i_t* as_lds_v2i(void* p) {
    return (lv2i_t*)(unsigned int)(unsigned long long)p;
}
#else
#define USE_ASYNC_LDS 0
#endif

// ---------------------------------------------------------------------------
// Kernel 1: per (b, kp) compute masked norms into LDS, iterative 10-smallest
// selection (tie-break lowest index, matching lax.top_k), gather candidates.
// One workgroup per (b,kp): 576 blocks x 256 threads.
// Fill phase uses gfx1250 async global->LDS copies (ASYNCcnt-pipelined,
// double buffered) when the toolchain exposes the builtins.
// ---------------------------------------------------------------------------
__global__ __launch_bounds__(256)
void topk_gather_kernel(const float* __restrict__ pcld,
                        const float* __restrict__ kpts,
                        const float* __restrict__ cpt,
                        const float* __restrict__ seg,
                        float* __restrict__ cans)
{
    __shared__ float vals[NPTS];
    __shared__ float rv[256];
    __shared__ int   ri[256];
    __shared__ int   sel[KCAND];
#if USE_ASYNC_LDS
    __shared__ float offbuf[2][CHUNK * 3];
    __shared__ float segbuf[2][CHUNK * 2];
#endif

    const int blk = blockIdx.x;
    const int b   = blk / NKP;
    const int kp  = blk % NKP;
    const int tid = threadIdx.x;

    // ---- Fill masked norms. seg argmax==1 iff seg[1] > seg[0]. ----
#if USE_ASYNC_LDS
    // Async pipeline: issue chunk c+1 while consuming chunk c. Each lane only
    // reads the LDS slots its own async loads wrote, and async loads complete
    // in issue order, so s_wait_asynccnt is the only sync required.
    {
        auto issue = [&](int c) {
            const int n = c * CHUNK + tid;
            const float* op = (kp < 8)
                ? (kpts + (((size_t)(b * NPTS + n) * 8 + kp) * 3))
                : (cpt  + ((size_t)(b * NPTS + n) * 3));
            const float* sp = seg + (size_t)(b * NPTS + n) * 2;
            const int buf = c & 1;
            // 3 x B32: same base pointers, imm offsets 0/4/8 advance both sides.
            __builtin_amdgcn_global_load_async_to_lds_b32(
                as_global_i32(op), as_lds_i32(&offbuf[buf][tid * 3]), 0, 0);
            __builtin_amdgcn_global_load_async_to_lds_b32(
                as_global_i32(op), as_lds_i32(&offbuf[buf][tid * 3]), 4, 0);
            __builtin_amdgcn_global_load_async_to_lds_b32(
                as_global_i32(op), as_lds_i32(&offbuf[buf][tid * 3]), 8, 0);
            __builtin_amdgcn_global_load_async_to_lds_b64(
                as_global_v2i(sp), as_lds_v2i(&segbuf[buf][tid * 2]), 0, 0);
        };
        issue(0);
        for (int c = 0; c < NCHUNK; ++c) {
            if (c + 1 < NCHUNK) {
                issue(c + 1);
                __builtin_amdgcn_s_wait_asynccnt(4);  // chunk c's 4 ops retired
            } else {
                __builtin_amdgcn_s_wait_asynccnt(0);
            }
            const int buf = c & 1;
            const float ox = offbuf[buf][tid * 3 + 0];
            const float oy = offbuf[buf][tid * 3 + 1];
            const float oz = offbuf[buf][tid * 3 + 2];
            const float s0 = segbuf[buf][tid * 2 + 0];
            const float s1 = segbuf[buf][tid * 2 + 1];
            const float m  = (s1 > s0) ? 1.0f : BIGF;
            vals[c * CHUNK + tid] = sqrtf(ox * ox + oy * oy + oz * oz) * m;
        }
    }
#else
    for (int n = tid; n < NPTS; n += 256) {
        const float* op = (kp < 8)
            ? (kpts + (((size_t)(b * NPTS + n) * 8 + kp) * 3))
            : (cpt  + ((size_t)(b * NPTS + n) * 3));
        __builtin_prefetch(op + 256 * 24, 0, 0);   // global_prefetch_b8
        float ox = op[0], oy = op[1], oz = op[2];
        float s0 = seg[(size_t)(b * NPTS + n) * 2 + 0];
        float s1 = seg[(size_t)(b * NPTS + n) * 2 + 1];
        float m  = (s1 > s0) ? 1.0f : BIGF;
        vals[n] = sqrtf(ox * ox + oy * oy + oz * oz) * m;
    }
#endif
    __syncthreads();

    // ---- 10 rounds of block-wide argmin over LDS. ----
    for (int it = 0; it < KCAND; ++it) {
        float v = 3.0e38f; int idx = 0x7fffffff;
        for (int n = tid; n < NPTS; n += 256) {
            float x = vals[n];
            if (x < v) { v = x; idx = n; }   // strict < keeps lowest n in-thread
        }
        rv[tid] = v; ri[tid] = idx;
        __syncthreads();
        for (int s = 128; s > 0; s >>= 1) {
            if (tid < s) {
                float v2 = rv[tid + s]; int i2 = ri[tid + s];
                if (v2 < rv[tid] || (v2 == rv[tid] && i2 < ri[tid])) {
                    rv[tid] = v2; ri[tid] = i2;
                }
            }
            __syncthreads();
        }
        if (tid == 0) {
            sel[it] = ri[0];
            vals[ri[0]] = 3.4e38f;
        }
        __syncthreads();
    }

    // ---- Gather: cand = pcld[idx] + off[idx] ----
    if (tid < KCAND) {
        const int n = sel[tid];
        const float* op = (kp < 8)
            ? (kpts + (((size_t)(b * NPTS + n) * 8 + kp) * 3))
            : (cpt  + ((size_t)(b * NPTS + n) * 3));
        const float* pp = pcld + (size_t)(b * NPTS + n) * 3;
        float* o = cans + ((size_t)(b * NKP + kp) * KCAND + tid) * 3;
        o[0] = pp[0] + op[0];
        o[1] = pp[1] + op[1];
        o[2] = pp[2] + op[2];
    }
}

// ---------------------------------------------------------------------------
// Kernel 2: clustering_with_std. One thread per (b,kp,dim) = 1728 groups of 10.
// Population std (ddof=0), keep |x-mu| < sigma*sd (sigma=1), count xf != 0.
// ---------------------------------------------------------------------------
__global__ __launch_bounds__(256)
void cluster_kernel(const float* __restrict__ cans,
                    float* __restrict__ kv,
                    float* __restrict__ out_kv)
{
    const int g = blockIdx.x * 256 + threadIdx.x;
    if (g >= BATCH * NKP * 3) return;
    const int b   = g / (NKP * 3);
    const int rem = g % (NKP * 3);
    const int kp  = rem / 3;
    const int d   = rem % 3;

    const float* base = cans + ((size_t)(b * NKP + kp) * KCAND) * 3 + d;
    float x[KCAND];
    float mu = 0.f;
    for (int i = 0; i < KCAND; ++i) { x[i] = base[i * 3]; mu += x[i]; }
    mu *= (1.0f / KCAND);
    float var = 0.f;
    for (int i = 0; i < KCAND; ++i) { float dd = x[i] - mu; var += dd * dd; }
    var *= (1.0f / KCAND);
    const float sd = sqrtf(var);

    float s = 0.f, nz = 0.f;
    for (int i = 0; i < KCAND; ++i) {
        const float xf = (fabsf(x[i] - mu) < sd) ? x[i] : 0.f;
        s  += xf;
        nz += (xf != 0.f) ? 1.f : 0.f;
    }
    const float r = s / fmaxf(nz, 1.f);
    kv[g]     = r;   // workspace copy for the Kabsch stage
    out_kv[g] = r;   // third tuple output
}

// ---------------------------------------------------------------------------
// Kernel 3: weighted Kabsch per batch. One wave32 per batch.
// H(3x3) = Am(3x9) * Bm(9x3) via a single v_wmma_f32_16x16x32_f16
// (M,N padded to 16, K=9 padded to 32 with zeros). Then Jacobi SVD of 3x3,
// det-sign fix on V[:,2], R = Vf U^T, t = cb - R ca.
// ---------------------------------------------------------------------------
__global__ __launch_bounds__(32)
void kabsch_wmma_kernel(const float* __restrict__ mesh,
                        const float* __restrict__ kv,
                        float* __restrict__ outR,
                        float* __restrict__ outT)
{
    const int b    = blockIdx.x;
    const int lane = threadIdx.x;
    const float* Mp = mesh + (size_t)b * NKP * 3;   // A points (9,3)
    const float* Bp = kv   + (size_t)b * NKP * 3;   // B points (9,3)

    // Centroids (w = 1, wsum = 9); redundant per lane, L2-cached, trivial.
    float ca[3], cb[3];
    for (int j = 0; j < 3; ++j) {
        float sa = 0.f, sb = 0.f;
        for (int n = 0; n < NKP; ++n) { sa += Mp[n * 3 + j]; sb += Bp[n * 3 + j]; }
        ca[j] = sa * (1.0f / NKP);
        cb[j] = sb * (1.0f / NKP);
    }

    // WMMA fragments per the 16-bit A 16x32 / B 32x16 lane layouts.
    // A: lane L (M=L%16, h=L/16), element e: K = (e<8) ? 8h+e : 16+8h+(e-8)
    // B: lane L (N=L%16, h=L/16), element e: K = (e<8) ? 8h+e : 16+8h+(e-8)
    const int Mrow = lane & 15;
    const int h    = lane >> 4;
    v16h af, bf;
    for (int e = 0; e < 16; ++e) {
        const int Ka = (e < 8) ? (8 * h + e) : (16 + 8 * h + (e - 8));
        const float av = (Mrow < 3 && Ka < NKP) ? (Mp[Ka * 3 + Mrow] - ca[Mrow]) : 0.f;
        af[e] = (_Float16)av;
        const int Kb = (e < 8) ? (8 * h + e) : (16 + 8 * h + (e - 8));
        const float bv = (Mrow < 3 && Kb < NKP) ? (Bp[Kb * 3 + Mrow] - cb[Mrow]) : 0.f;
        bf[e] = (_Float16)bv;
    }
    v8f c = {};
    // D = A x B + C : H[i][j] lands in accumulator VGPR i, lane j.
    c = __builtin_amdgcn_wmma_f32_16x16x32_f16(
        /*neg_a=*/false, af, /*neg_b=*/false, bf,
        /*c_mod=*/(short)0, c, /*reuse_a=*/false, /*reuse_b=*/false);

    float H[3][3];
    for (int i = 0; i < 3; ++i)
        for (int j = 0; j < 3; ++j)
            H[i][j] = __shfl(c[i], j, 32);

    // ---- 3x3 SVD via cyclic Jacobi on S = H^T H (all lanes redundantly) ----
    float A[3][3], V[3][3];
    for (int i = 0; i < 3; ++i)
        for (int j = 0; j < 3; ++j) {
            float s = 0.f;
            for (int k = 0; k < 3; ++k) s += H[k][i] * H[k][j];
            A[i][j] = s;
            V[i][j] = (i == j) ? 1.f : 0.f;
        }

    for (int sweep = 0; sweep < 10; ++sweep) {
        for (int pair = 0; pair < 3; ++pair) {
            const int p = (pair == 2) ? 1 : 0;
            const int q = (pair == 0) ? 1 : 2;
            const float apq = A[p][q];
            if (fabsf(apq) < 1e-30f) continue;
            const float theta = (A[q][q] - A[p][p]) / (2.f * apq);
            const float t = ((theta >= 0.f) ? 1.f : -1.f) /
                            (fabsf(theta) + sqrtf(theta * theta + 1.f));
            const float cth = 1.f / sqrtf(t * t + 1.f);
            const float sth = t * cth;
            for (int k = 0; k < 3; ++k) {       // A = A*J (columns p,q)
                const float akp = A[k][p], akq = A[k][q];
                A[k][p] = cth * akp - sth * akq;
                A[k][q] = sth * akp + cth * akq;
            }
            for (int k = 0; k < 3; ++k) {       // A = J^T*A (rows p,q)
                const float apk = A[p][k], aqk = A[q][k];
                A[p][k] = cth * apk - sth * aqk;
                A[q][k] = sth * apk + cth * aqk;
            }
            for (int k = 0; k < 3; ++k) {       // V = V*J
                const float vkp = V[k][p], vkq = V[k][q];
                V[k][p] = cth * vkp - sth * vkq;
                V[k][q] = sth * vkp + cth * vkq;
            }
        }
    }

    // Sort eigenpairs descending (column 2 <-> smallest singular value).
    float lam[3] = { A[0][0], A[1][1], A[2][2] };
    for (int i = 0; i < 2; ++i)
        for (int j = 0; j < 2 - i; ++j)
            if (lam[j] < lam[j + 1]) {
                const float tl = lam[j]; lam[j] = lam[j + 1]; lam[j + 1] = tl;
                for (int k = 0; k < 3; ++k) {
                    const float tv = V[k][j]; V[k][j] = V[k][j + 1]; V[k][j + 1] = tv;
                }
            }

    // U columns: u_i = H v_i / s_i ; rebuild degenerate smallest via cross.
    float U[3][3], sv[3];
    for (int i = 0; i < 3; ++i) {
        sv[i] = sqrtf(fmaxf(lam[i], 0.f));
        for (int r = 0; r < 3; ++r) {
            float s = 0.f;
            for (int k = 0; k < 3; ++k) s += H[r][k] * V[k][i];
            U[r][i] = (sv[i] > 1e-12f) ? (s / sv[i]) : s;
        }
    }
    if (sv[2] <= 1e-12f) {
        U[0][2] = U[1][0] * U[2][1] - U[2][0] * U[1][1];
        U[1][2] = U[2][0] * U[0][1] - U[0][0] * U[2][1];
        U[2][2] = U[0][0] * U[1][1] - U[1][0] * U[0][1];
    }

    // R0 = V U^T, d = sign(det R0), flip V[:,2], R = Vf U^T, t = cb - R ca.
    float R0[3][3];
    for (int i = 0; i < 3; ++i)
        for (int j = 0; j < 3; ++j) {
            float s = 0.f;
            for (int k = 0; k < 3; ++k) s += V[i][k] * U[j][k];
            R0[i][j] = s;
        }
    const float det =
        R0[0][0] * (R0[1][1] * R0[2][2] - R0[1][2] * R0[2][1]) -
        R0[0][1] * (R0[1][0] * R0[2][2] - R0[1][2] * R0[2][0]) +
        R0[0][2] * (R0[1][0] * R0[2][1] - R0[1][1] * R0[2][0]);
    const float dsg = (det > 0.f) ? 1.f : ((det < 0.f) ? -1.f : 0.f);
    for (int k = 0; k < 3; ++k) V[k][2] *= dsg;

    float R[3][3];
    for (int i = 0; i < 3; ++i)
        for (int j = 0; j < 3; ++j) {
            float s = 0.f;
            for (int k = 0; k < 3; ++k) s += V[i][k] * U[j][k];
            R[i][j] = s;
        }
    float tvec[3];
    for (int j = 0; j < 3; ++j) {
        float s = 0.f;
        for (int k = 0; k < 3; ++k) s += R[j][k] * ca[k];
        tvec[j] = cb[j] - s;
    }

    if (lane == 0) {
        for (int i = 0; i < 3; ++i)
            for (int j = 0; j < 3; ++j)
                outR[(size_t)b * 9 + i * 3 + j] = R[i][j];
        for (int j = 0; j < 3; ++j)
            outT[(size_t)b * 3 + j] = tvec[j];
    }
}

// ---------------------------------------------------------------------------
// Launch: out layout = [R (64*3*3=576) | t (64*3=192) | kpts_voted (64*9*3)]
// ---------------------------------------------------------------------------
extern "C" void kernel_launch(void* const* d_in, const int* in_sizes, int n_in,
                              void* d_out, int out_size, void* d_ws, size_t ws_size,
                              hipStream_t stream) {
    const float* pcld = (const float*)d_in[0];
    const float* kpts = (const float*)d_in[1];
    const float* cpt  = (const float*)d_in[2];
    const float* seg  = (const float*)d_in[3];
    const float* mesh = (const float*)d_in[4];
    float* out = (float*)d_out;

    float* cans = (float*)d_ws;                               // 64*9*10*3 floats
    float* kv   = cans + (size_t)BATCH * NKP * KCAND * 3;     // 64*9*3 floats

    topk_gather_kernel<<<BATCH * NKP, 256, 0, stream>>>(pcld, kpts, cpt, seg, cans);
    cluster_kernel<<<(BATCH * NKP * 3 + 255) / 256, 256, 0, stream>>>(
        cans, kv, out + 576 + 192);
    kabsch_wmma_kernel<<<BATCH, 32, 0, stream>>>(mesh, kv, out, out + 576);
}
